// AutoCorrelation_90245852824063
// MI455X (gfx1250) — matrix operand order repaired and checked by reading, not compiled, not run on hardware
//
#include <hip/hip_runtime.h>
#include <math.h>

// Problem constants (from reference): x [B,T,D], H heads, hd = D/H
constexpr int B_  = 4;
constexpr int T_  = 2048;
constexpr int D_  = 1024;
constexpr int H_  = 16;
constexpr int HD_ = 64;          // head dim
constexpr int QB_ = 128;         // query rows per workgroup (8 waves x 16 rows)
constexpr int KB_ = 64;          // keys per inner iteration
constexpr int LP_ = 80;          // LDS row pitch in bf16 elems (160B: 16B multiple, bank-spread)

typedef __attribute__((ext_vector_type(16))) __bf16 v16bf;
typedef __attribute__((ext_vector_type(8)))  __bf16 v8bf;
typedef __attribute__((ext_vector_type(8)))  float  v8f;
typedef __attribute__((ext_vector_type(4)))  int    v4i;

#if defined(__HIP_DEVICE_COMPILE__) && __has_builtin(__builtin_amdgcn_global_load_async_to_lds_b128)
#define ASYNC_TILE_LOADS 1
#else
#define ASYNC_TILE_LOADS 0
#endif

__device__ __forceinline__ void wait_async_le4() {
#if ASYNC_TILE_LOADS
#if __has_builtin(__builtin_amdgcn_s_wait_asynccnt)
    __builtin_amdgcn_s_wait_asynccnt(4);
#else
    asm volatile("s_wait_asynccnt 0x4" ::: "memory");
#endif
#endif
}
__device__ __forceinline__ void wait_async_le0() {
#if ASYNC_TILE_LOADS
#if __has_builtin(__builtin_amdgcn_s_wait_asynccnt)
    __builtin_amdgcn_s_wait_asynccnt(0);
#else
    asm volatile("s_wait_asynccnt 0x0" ::: "memory");
#endif
#endif
}

// ---------------------------------------------------------------------------
// Kernel 1: column means over T for each (b, d) -> mean[b*D + d]
// ---------------------------------------------------------------------------
__global__ __launch_bounds__(256)
void colmean_kernel(const float* __restrict__ x, float* __restrict__ mean) {
    __shared__ float red[256];
    const int b    = blockIdx.x / (D_ / 64);
    const int dblk = blockIdx.x % (D_ / 64);
    const int dcol = dblk * 64 + (threadIdx.x & 63);
    const int tseg = threadIdx.x >> 6;            // 0..3
    const float* p = x + (size_t)b * T_ * D_ + dcol;
    float s = 0.0f;
    const int t0 = tseg * (T_ / 4), t1 = t0 + (T_ / 4);
    for (int t = t0; t < t1; ++t) s += p[(size_t)t * D_];
    red[threadIdx.x] = s;
    __syncthreads();
    if (tseg == 0) {
        s = red[threadIdx.x] + red[threadIdx.x + 64] +
            red[threadIdx.x + 128] + red[threadIdx.x + 192];
        mean[b * D_ + dcol] = s * (1.0f / (float)T_);
    }
}

// ---------------------------------------------------------------------------
// Kernel 2: pack pass. Kc[b,h,t,hd] = bf16(x - mean)  (centered, head-major)
//           Vt[b,h,hd,t] = bf16(x)                    (uncentered, transposed)
// grid = B*H*(T/64), 256 threads.
// ---------------------------------------------------------------------------
__global__ __launch_bounds__(256)
void pack_kernel(const float* __restrict__ x, const float* __restrict__ mean,
                 __bf16* __restrict__ Kc, __bf16* __restrict__ Vt) {
    __shared__ __bf16 Lt[HD_][72];
    const int ntb  = T_ / 64;
    const int bh   = blockIdx.x / ntb;
    const int tblk = blockIdx.x % ntb;
    const int b = bh / H_, h = bh % H_;
    const float* xb = x    + (size_t)b * T_ * D_ + h * HD_;
    const float* mp = mean + (size_t)b * D_      + h * HD_;
    __bf16* kcB = Kc + ((size_t)bh * T_ + tblk * 64) * HD_;
    __bf16* vtB = Vt + (size_t)bh * HD_ * T_ + tblk * 64;
    #pragma unroll
    for (int i = 0; i < 16; ++i) {
        const int idx = threadIdx.x + i * 256;
        const int tt = idx >> 6, dd = idx & 63;
        const float v = xb[(size_t)(tblk * 64 + tt) * D_ + dd];
        kcB[tt * HD_ + dd] = (__bf16)(v - mp[dd]);
        Lt[dd][tt] = (__bf16)v;
    }
    __syncthreads();
    #pragma unroll
    for (int i = 0; i < 16; ++i) {
        const int idx = threadIdx.x + i * 256;
        const int dd = idx >> 6, tt = idx & 63;
        vtB[(size_t)dd * T_ + tt] = Lt[dd][tt];
    }
}

// ---------------------------------------------------------------------------
// Shared per-tile flash-attention step: S = Q*K^T (2 WMMA x 4 tiles), online
// softmax, P staged via LDS, O += P*V (2 WMMA x 4 tiles). C-layout: VGPR slot
// r = row r(+8 for lanes 16-31), lane%16 = column.
// ---------------------------------------------------------------------------
__device__ __forceinline__ void attn_tile_step(
    const __bf16* __restrict__ Ks,   // [64][LP_] centered keys, [key][hd]
    const __bf16* __restrict__ Vt,   // [64][LP_] values transposed, [hd][key]
    __bf16* __restrict__ PsW,        // [16][LP_] wave-private P staging
    const v16bf qa[2], v8f o[4], float m_run[8], float l_run[8],
    const int lrow, const int half, const int kb8, const int roff) {

    v8f s[4];
    #pragma unroll
    for (int t = 0; t < 4; ++t) {
        v16bf kbf[2];
        #pragma unroll
        for (int f = 0; f < 2; ++f) {
            const __bf16* src = Ks + (t * 16 + lrow) * LP_ + f * 32 + half * 8;
            #pragma unroll
            for (int e = 0; e < 16; ++e) kbf[f][e] = src[(e < 8) ? e : (e + 8)];
        }
        v8f c;
        #pragma unroll
        for (int j = 0; j < 8; ++j) c[j] = 0.0f;
        c = __builtin_amdgcn_wmma_f32_16x16x32_bf16(false, qa[0], false, kbf[0],
                                                    (short)0, c, false, false);
        c = __builtin_amdgcn_wmma_f32_16x16x32_bf16(false, qa[1], false, kbf[1],
                                                    (short)0, c, false, false);
        s[t] = c;
    }

    float mnew[8];
    #pragma unroll
    for (int r = 0; r < 8; ++r) mnew[r] = m_run[r];
    #pragma unroll
    for (int t = 0; t < 4; ++t)
        #pragma unroll
        for (int r = 0; r < 8; ++r) mnew[r] = fmaxf(mnew[r], s[t][r]);
    #pragma unroll
    for (int off = 1; off < 16; off <<= 1)
        #pragma unroll
        for (int r = 0; r < 8; ++r)
            mnew[r] = fmaxf(mnew[r], __shfl_xor(mnew[r], off, 32));

    float rs[8];
    #pragma unroll
    for (int r = 0; r < 8; ++r) rs[r] = 0.0f;
    #pragma unroll
    for (int t = 0; t < 4; ++t) {
        #pragma unroll
        for (int r = 0; r < 8; ++r) {
            const float p = __expf(s[t][r] - mnew[r]);
            rs[r] += p;
            PsW[(roff + r) * LP_ + t * 16 + lrow] = (__bf16)p;
        }
    }
    #pragma unroll
    for (int off = 1; off < 16; off <<= 1)
        #pragma unroll
        for (int r = 0; r < 8; ++r) rs[r] += __shfl_xor(rs[r], off, 32);

    #pragma unroll
    for (int r = 0; r < 8; ++r) {
        const float sc = __expf(m_run[r] - mnew[r]);
        l_run[r] = l_run[r] * sc + rs[r];
        m_run[r] = mnew[r];
        #pragma unroll
        for (int t = 0; t < 4; ++t) o[t][r] *= sc;
    }

    v16bf pa[2];
    #pragma unroll
    for (int f = 0; f < 2; ++f) {
        #pragma unroll
        for (int e = 0; e < 16; ++e) {
            const int k = f * 32 + kb8 + ((e < 8) ? e : (e + 8));
            pa[f][e] = PsW[lrow * LP_ + k];
        }
    }
    #pragma unroll
    for (int t = 0; t < 4; ++t) {
        v16bf vbf[2];
        #pragma unroll
        for (int f = 0; f < 2; ++f) {
            const __bf16* src = Vt + (t * 16 + lrow) * LP_ + f * 32 + half * 8;
            #pragma unroll
            for (int e = 0; e < 16; ++e) vbf[f][e] = src[(e < 8) ? e : (e + 8)];
        }
        o[t] = __builtin_amdgcn_wmma_f32_16x16x32_bf16(false, pa[0], false, vbf[0],
                                                       (short)0, o[t], false, false);
        o[t] = __builtin_amdgcn_wmma_f32_16x16x32_bf16(false, pa[1], false, vbf[1],
                                                       (short)0, o[t], false, false);
    }
}

// ---------------------------------------------------------------------------
// Tile fill: 16B chunks, async-to-LDS when available (double-buffer friendly).
// K tile: 8KB fully contiguous in Kc. V tile: 64 rows of 128B, row stride T*2.
// ---------------------------------------------------------------------------
__device__ __forceinline__ void issue_tile_loads(
    const __bf16* __restrict__ kSrc, const __bf16* __restrict__ vSrc,
    __bf16* kDst, __bf16* vDst, const int tid) {
    #pragma unroll
    for (int i = 0; i < 2; ++i) {
        const int c = tid + i * 256;          // 0..511 chunk id
        const int row = c >> 3, col = c & 7;
        const char* gk = (const char*)kSrc + (size_t)c * 16;
        const char* gv = (const char*)vSrc + (size_t)row * (T_ * 2) + col * 16;
        char* lk = (char*)kDst + row * (LP_ * 2) + col * 16;
        char* lv = (char*)vDst + row * (LP_ * 2) + col * 16;
#if ASYNC_TILE_LOADS
        __builtin_amdgcn_global_load_async_to_lds_b128(
            (__attribute__((address_space(1))) v4i*)gk,
            (__attribute__((address_space(3))) v4i*)lk, 0, 0);
        __builtin_amdgcn_global_load_async_to_lds_b128(
            (__attribute__((address_space(1))) v4i*)gv,
            (__attribute__((address_space(3))) v4i*)lv, 0, 0);
#else
        *(v8bf*)lk = *(const v8bf*)gk;
        *(v8bf*)lv = *(const v8bf*)gv;
#endif
    }
}

// ---------------------------------------------------------------------------
// Kernel 3a: flash attention on packed bf16 inputs, double-buffered async LDS.
// ---------------------------------------------------------------------------
__global__ __launch_bounds__(256)
void flash_attn_packed_kernel(const __bf16* __restrict__ Kc,
                              const __bf16* __restrict__ Vt,
                              float* __restrict__ out) {
    __shared__ __align__(16) __bf16 Kbuf[2][KB_][LP_];
    __shared__ __align__(16) __bf16 Vbuf[2][HD_][LP_];
    __shared__ __align__(16) __bf16 Ps[8][16][LP_];

    const int nq   = T_ / QB_;
    const int bh   = blockIdx.x / nq;
    const int qblk = blockIdx.x % nq;
    const int b = bh / H_, h = bh % H_;

    const int tid  = threadIdx.x;
    const int w    = tid >> 5, lane = tid & 31;
    const int lrow = lane & 15;
    const int half = lane >> 4;
    const int kb8  = half * 8;
    const int roff = half * 8;

    const __bf16* kcBase = Kc + (size_t)bh * T_ * HD_;
    const __bf16* vtBase = Vt + (size_t)bh * HD_ * T_;

    // ---- Q A-fragments straight from packed centered-bf16 ------------------
    const int qrow = qblk * QB_ + w * 16 + lrow;
    const __bf16* qb = kcBase + (size_t)qrow * HD_;
    v16bf qa[2];
    #pragma unroll
    for (int f = 0; f < 2; ++f) {
        const v8bf lo = *(const v8bf*)(qb + f * 32 + kb8);
        const v8bf hi = *(const v8bf*)(qb + f * 32 + kb8 + 16);
        #pragma unroll
        for (int e = 0; e < 8; ++e) { qa[f][e] = lo[e]; qa[f][8 + e] = hi[e]; }
    }

    v8f o[4];
    #pragma unroll
    for (int t = 0; t < 4; ++t)
        #pragma unroll
        for (int j = 0; j < 8; ++j) o[t][j] = 0.0f;
    float m_run[8], l_run[8];
    #pragma unroll
    for (int r = 0; r < 8; ++r) { m_run[r] = -3.0e38f; l_run[r] = 0.0f; }

    constexpr int NKB = T_ / KB_;
    issue_tile_loads(kcBase, vtBase, &Kbuf[0][0][0], &Vbuf[0][0][0], tid);

    for (int kblk = 0; kblk < NKB; ++kblk) {
        const int cur = kblk & 1, nxt = cur ^ 1;
        if (kblk + 1 < NKB) {
            issue_tile_loads(kcBase + (size_t)(kblk + 1) * KB_ * HD_,
                             vtBase + (kblk + 1) * KB_,
                             &Kbuf[nxt][0][0], &Vbuf[nxt][0][0], tid);
            wait_async_le4();              // current tile (issued last iter) done
        } else {
            wait_async_le0();
        }
        __syncthreads();                   // all waves see current tile

        attn_tile_step(&Kbuf[cur][0][0], &Vbuf[cur][0][0], &Ps[w][0][0],
                       qa, o, m_run, l_run, lrow, half, kb8, roff);

        __syncthreads();                   // tile consumed before refill
    }

    float* op = out + (size_t)b * T_ * D_ + h * HD_;
    #pragma unroll
    for (int r = 0; r < 8; ++r) {
        const float inv = 1.0f / l_run[r];
        const int row = qblk * QB_ + w * 16 + roff + r;
        #pragma unroll
        for (int t = 0; t < 4; ++t)
            op[(size_t)row * D_ + t * 16 + lrow] = o[t][r] * inv;
    }
}

// ---------------------------------------------------------------------------
// Kernel 3b: fallback (small scratch): convert fp32 tiles in-kernel (round-1).
// ---------------------------------------------------------------------------
__global__ __launch_bounds__(256)
void flash_attn_legacy_kernel(const float* __restrict__ x,
                              const float* __restrict__ mean,
                              float* __restrict__ out) {
    __shared__ __align__(16) __bf16 Ks[KB_][LP_];
    __shared__ __align__(16) __bf16 Vt[HD_][LP_];
    __shared__ __align__(16) __bf16 Ps[8][16][LP_];

    const int nq   = T_ / QB_;
    const int bh   = blockIdx.x / nq;
    const int qblk = blockIdx.x % nq;
    const int b = bh / H_, h = bh % H_;
    const int w = threadIdx.x >> 5, lane = threadIdx.x & 31;
    const int lrow = lane & 15, half = lane >> 4;
    const int kb8 = half * 8, roff = half * 8;

    const float* xb = x    + (size_t)b * T_ * D_ + h * HD_;
    const float* mp = mean + (size_t)b * D_      + h * HD_;

    const int qrow = qblk * QB_ + w * 16 + lrow;
    const float* qp = xb + (size_t)qrow * D_;
    v16bf qa[2];
    #pragma unroll
    for (int f = 0; f < 2; ++f)
        #pragma unroll
        for (int e = 0; e < 16; ++e) {
            const int d = f * 32 + kb8 + ((e < 8) ? e : (e + 8));
            qa[f][e] = (__bf16)(qp[d] - mp[d]);
        }

    v8f o[4];
    #pragma unroll
    for (int t = 0; t < 4; ++t)
        #pragma unroll
        for (int j = 0; j < 8; ++j) o[t][j] = 0.0f;
    float m_run[8], l_run[8];
    #pragma unroll
    for (int r = 0; r < 8; ++r) { m_run[r] = -3.0e38f; l_run[r] = 0.0f; }

    for (int kblk = 0; kblk < T_ / KB_; ++kblk) {
        __syncthreads();
        const float* kp = xb + (size_t)(kblk * KB_) * D_;
        for (int idx = threadIdx.x; idx < KB_ * HD_; idx += 256) {
            const int kk = idx >> 6, dd = idx & 63;
            const float v = kp[(size_t)kk * D_ + dd];
            Ks[kk][dd] = (__bf16)(v - mp[dd]);
            Vt[dd][kk] = (__bf16)v;
        }
        __syncthreads();
        attn_tile_step(&Ks[0][0], &Vt[0][0], &Ps[w][0][0],
                       qa, o, m_run, l_run, lrow, half, kb8, roff);
    }

    float* op = out + (size_t)b * T_ * D_ + h * HD_;
    #pragma unroll
    for (int r = 0; r < 8; ++r) {
        const float inv = 1.0f / l_run[r];
        const int row = qblk * QB_ + w * 16 + roff + r;
        #pragma unroll
        for (int t = 0; t < 4; ++t)
            op[(size_t)row * D_ + t * 16 + lrow] = o[t][r] * inv;
    }
}

// ---------------------------------------------------------------------------
extern "C" void kernel_launch(void* const* d_in, const int* in_sizes, int n_in,
                              void* d_out, int out_size, void* d_ws, size_t ws_size,
                              hipStream_t stream) {
    const float* x = (const float*)d_in[0];
    float* out = (float*)d_out;
    char* ws = (char*)d_ws;

    float* mean = (float*)ws;                       // B*D f32 = 16 KB
    const size_t kcOff = 16384;
    const size_t bhTail = (size_t)B_ * H_ * T_ * HD_ * sizeof(__bf16);  // 16 MB
    const size_t vtOff = kcOff + bhTail;
    const size_t need  = vtOff + bhTail;            // ~32 MB + 16 KB

    colmean_kernel<<<dim3(B_ * (D_ / 64)), dim3(256), 0, stream>>>(x, mean);

    if (ws_size >= need) {
        __bf16* Kc = (__bf16*)(ws + kcOff);
        __bf16* Vt = (__bf16*)(ws + vtOff);
        pack_kernel<<<dim3(B_ * H_ * (T_ / 64)), dim3(256), 0, stream>>>(x, mean, Kc, Vt);
        flash_attn_packed_kernel<<<dim3(B_ * H_ * (T_ / QB_)), dim3(256), 0, stream>>>(Kc, Vt, out);
    } else {
        flash_attn_legacy_kernel<<<dim3(B_ * H_ * (T_ / QB_)), dim3(256), 0, stream>>>(x, mean, out);
    }
}
